// CustomMultiheadAttention_40450001994243
// MI455X (gfx1250) — matrix operand order repaired and checked by reading, not compiled, not run on hardware
//
#include <hip/hip_runtime.h>
#include <math.h>

// ---------------------------------------------------------------------------
// CDNA5 (gfx1250) wave32 WMMA bf16 pipeline for low-rank linear attention.
// v_wmma_f32_16x16x32_bf16 everywhere; K-step 64; double-buffered LDS;
// 8 WMMAs per barrier in the main GEMM.
// ---------------------------------------------------------------------------

typedef __bf16 bf16;
typedef __attribute__((ext_vector_type(16))) __bf16 v16bf;
typedef __attribute__((ext_vector_type(8)))  float  v8f;

union BF16Frag { v16bf v; uint4 u[2]; };
union H8u      { uint4 u; bf16 h[8]; };

#define WMMA_BF16(A, B, C) \
  __builtin_amdgcn_wmma_f32_16x16x32_bf16(false, (A), false, (B), (short)0, (C), false, false)

// A-fragment (16x32 bf16, M x K): lane m = lane&15; lanes 0-15 hold K=0..7 &
// 16..23, lanes 16-31 hold K=8..15 & 24..31 (ISA 7.12.2).  `row` points at the
// first of 32 contiguous bf16 covering this lane's M row, K window.
__device__ __forceinline__ v16bf make_a_frag(const bf16* row, int lane) {
  BF16Frag f;
  const uint4* p = (const uint4*)row;   // chunks of 8 bf16
  int c = lane >> 4;                    // K base 0 or 8
  f.u[0] = p[c];
  f.u[1] = p[c + 2];
  return f.v;
}

// B-fragment (32x16 bf16, K x N): lane n = lane&15; lanes 0-15 hold K=0..7 &
// 16..23, lanes 16-31 hold K=8..15 & 24..31.  Tile staged N-major so K is contiguous.
__device__ __forceinline__ v16bf make_b_frag(const bf16* row, int lane) {
  BF16Frag f;
  const uint4* p = (const uint4*)row;
  int c = lane >> 4;                    // K base 0 or 8
  f.u[0] = p[c];
  f.u[1] = p[c + 2];
  return f.v;
}

__device__ __forceinline__ void cvt_store8(bf16* d, float4 f0, float4 f1) {
  d[0] = (bf16)f0.x; d[1] = (bf16)f0.y; d[2] = (bf16)f0.z; d[3] = (bf16)f0.w;
  d[4] = (bf16)f1.x; d[5] = (bf16)f1.y; d[6] = (bf16)f1.z; d[7] = (bf16)f1.w;
}

// ---------------------------------------------------------------------------
// Generic GEMM:  C[M,N] = act( A[M,K] * W[N,K]^T + bias )
//   A: f32 or bf16 (converted while staging); W: f32 (converted while staging)
//   ACT: 0 = none, 1 = elu(x)+1  (= x+1 if x>0 else exp(x))
// 256 threads = 8 waves; block tile 128(M) x 64(N); wave tile 32x32;
// K-step 64; double-buffered LDS (48KB); one barrier per 8 WMMAs.
// ---------------------------------------------------------------------------
template <bool A_F32, bool HAS_BIAS, int ACT, bool OUT_BF16>
__global__ __launch_bounds__(256) void gemm_wmma_kernel(
    const void* __restrict__ Aptr, const float* __restrict__ Wptr,
    const float* __restrict__ bias, void* __restrict__ Cptr,
    int M, int N, int K) {
  __shared__ __align__(16) bf16 Abuf[2][128][64];   // 2 x 16KB
  __shared__ __align__(16) bf16 Bbuf[2][64][64];    // 2 x 8KB, [n][k]

  const int tid  = threadIdx.x;
  const int lane = tid & 31;
  const int wave = tid >> 5;
  const int wr = (wave >> 1) * 32;   // 0,32,64,96
  const int wc = (wave & 1) * 32;    // 0,32
  const int rowBase = blockIdx.x * 128;
  const int colBase = blockIdx.y * 64;

  // staging: A 128x64 (32 elems/thread), B 64x64 (16 elems/thread)
  const int arow = tid >> 1;          // 0..127
  const int acol = (tid & 1) * 32;    // 0,32
  const int brow = tid >> 2;          // 0..63
  const int bcol = (tid & 3) * 16;    // 0,16,32,48

  const float* Af = (const float*)Aptr;
  const bf16*  Ah = (const bf16*)Aptr;
  const size_t aoff = (size_t)(rowBase + arow) * K + acol;
  const size_t woff = (size_t)(colBase + brow) * K + bcol;

  float4 fa[8]; uint4 ha[4]; float4 fw[4];

  auto loadRegs = [&](int k) {
    if (A_F32) {
      const float4* p = (const float4*)(Af + aoff + k);
#pragma unroll
      for (int i = 0; i < 8; ++i) fa[i] = p[i];
    } else {
      const uint4* p = (const uint4*)(Ah + aoff + k);
#pragma unroll
      for (int i = 0; i < 4; ++i) ha[i] = p[i];
    }
    const float4* p = (const float4*)(Wptr + woff + k);
#pragma unroll
    for (int i = 0; i < 4; ++i) fw[i] = p[i];
  };
  auto storeRegs = [&](int nb) {
    if (A_F32) {
#pragma unroll
      for (int i = 0; i < 4; ++i)
        cvt_store8(&Abuf[nb][arow][acol + 8 * i], fa[2 * i], fa[2 * i + 1]);
    } else {
      uint4* d = (uint4*)&Abuf[nb][arow][acol];
#pragma unroll
      for (int i = 0; i < 4; ++i) d[i] = ha[i];
    }
#pragma unroll
    for (int i = 0; i < 2; ++i)
      cvt_store8(&Bbuf[nb][brow][bcol + 8 * i], fw[2 * i], fw[2 * i + 1]);
  };

  v8f acc00 = {}, acc01 = {}, acc10 = {}, acc11 = {};

  loadRegs(0);
  storeRegs(0);
  __syncthreads();

  int buf = 0;
  for (int k0 = 0; k0 < K; k0 += 64) {
    const bool pre = (k0 + 64) < K;
    if (pre) {
      loadRegs(k0 + 64);
      if (k0 + 128 < K) {
        if (A_F32) __builtin_prefetch(Af + aoff + k0 + 128, 0, 1);
        else       __builtin_prefetch(Ah + aoff + k0 + 128, 0, 1);
      }
    }

    // ---- 8 WMMAs from current buffer ----
#pragma unroll
    for (int kk = 0; kk < 64; kk += 32) {
      v16bf A0 = make_a_frag(&Abuf[buf][wr +      (lane & 15)][kk], lane);
      v16bf A1 = make_a_frag(&Abuf[buf][wr + 16 + (lane & 15)][kk], lane);
      v16bf B0 = make_b_frag(&Bbuf[buf][wc +      (lane & 15)][kk], lane);
      v16bf B1 = make_b_frag(&Bbuf[buf][wc + 16 + (lane & 15)][kk], lane);
      acc00 = WMMA_BF16(A0, B0, acc00);
      acc01 = WMMA_BF16(A0, B1, acc01);
      acc10 = WMMA_BF16(A1, B0, acc10);
      acc11 = WMMA_BF16(A1, B1, acc11);
    }

    if (pre) storeRegs(buf ^ 1);
    __syncthreads();
    buf ^= 1;
  }

  // ---- epilogue ----
  const int n0 = colBase + wc + (lane & 15);
  const int n1 = n0 + 16;
  const float bv0 = HAS_BIAS ? bias[n0] : 0.0f;
  const float bv1 = HAS_BIAS ? bias[n1] : 0.0f;
#pragma unroll
  for (int v = 0; v < 8; ++v) {
    const int m0 = rowBase + wr + v + 8 * (lane >> 4);
    const int m1 = m0 + 16;
    float v00 = acc00[v] + bv0, v01 = acc01[v] + bv1;
    float v10 = acc10[v] + bv0, v11 = acc11[v] + bv1;
    if (ACT == 1) {
      v00 = (v00 > 0.0f) ? (v00 + 1.0f) : __expf(v00);
      v01 = (v01 > 0.0f) ? (v01 + 1.0f) : __expf(v01);
      v10 = (v10 > 0.0f) ? (v10 + 1.0f) : __expf(v10);
      v11 = (v11 > 0.0f) ? (v11 + 1.0f) : __expf(v11);
    }
    if (OUT_BF16) {
      bf16* C = (bf16*)Cptr;
      C[(size_t)m0 * N + n0] = (bf16)v00;
      C[(size_t)m0 * N + n1] = (bf16)v01;
      C[(size_t)m1 * N + n0] = (bf16)v10;
      C[(size_t)m1 * N + n1] = (bf16)v11;
    } else {
      float* C = (float*)Cptr;
      C[(size_t)m0 * N + n0] = v00;
      C[(size_t)m0 * N + n1] = v01;
      C[(size_t)m1 * N + n0] = v10;
      C[(size_t)m1 * N + n1] = v11;
    }
  }
}

// ---------------------------------------------------------------------------
// kv partials: for (bh, schunk) accumulate kv[d][e] = sum K[n][d]*V[n][e]
// over n in [sc*512, sc*512+512).  Double-buffered, 64 seq rows per stage,
// 2 WMMAs per barrier.  Writes coalesced f32 partial kv[d][e].
// 512 threads = 16 waves, each owns one 16x16 tile of the 64x64 output.
// ---------------------------------------------------------------------------
__global__ __launch_bounds__(512) void kv_kernel(
    const bf16* __restrict__ Kf, const bf16* __restrict__ Vf,
    float* __restrict__ kvPart) {
  __shared__ __align__(16) bf16 Kt[2][64][64];  // [d][n], 2 x 8KB
  __shared__ __align__(16) bf16 Vt[2][64][64];  // [e][n], 2 x 8KB
  const int bh = blockIdx.x;                 // 0..63
  const int sc = blockIdx.y;                 // 0..7
  const int b = bh >> 4, h = bh & 15;
  const size_t base = ((size_t)b * 4096 + sc * 512) * 1024 + (size_t)h * 64;
  const int tid = threadIdx.x;
  const int lane = tid & 31;
  const int wave = tid >> 5;                 // 0..15
  const int wr = (wave >> 2) * 16;           // d-tile
  const int wc = (wave & 3) * 16;            // e-tile

  const int sn = tid >> 3;                   // 0..63 (seq row in chunk)
  const int sd = (tid & 7) * 8;              // 0..56 (channel)

  H8u k8, v8;
  auto loadRegs = [&](int s0) {
    k8.u = *(const uint4*)(Kf + base + (size_t)(s0 + sn) * 1024 + sd);
    v8.u = *(const uint4*)(Vf + base + (size_t)(s0 + sn) * 1024 + sd);
  };
  auto storeRegs = [&](int nb) {
#pragma unroll
    for (int j = 0; j < 8; ++j) {
      Kt[nb][sd + j][sn] = k8.h[j];
      Vt[nb][sd + j][sn] = v8.h[j];
    }
  };

  v8f acc = {};

  loadRegs(0);
  storeRegs(0);
  __syncthreads();

  int buf = 0;
  for (int s0 = 0; s0 < 512; s0 += 64) {
    const bool pre = (s0 + 64) < 512;
    if (pre) loadRegs(s0 + 64);
#pragma unroll
    for (int kk = 0; kk < 64; kk += 32) {
      v16bf af = make_a_frag(&Kt[buf][wr + (lane & 15)][kk], lane);
      v16bf bf = make_b_frag(&Vt[buf][wc + (lane & 15)][kk], lane);
      acc = WMMA_BF16(af, bf, acc);
    }
    if (pre) storeRegs(buf ^ 1);
    __syncthreads();
    buf ^= 1;
  }

  // coalesced partial store, [d][e] order
  float* out = kvPart + ((size_t)bh * 8 + sc) * 4096;
  const int e = wc + (lane & 15);
#pragma unroll
  for (int v = 0; v < 8; ++v) {
    int d = wr + v + 8 * (lane >> 4);
    out[(size_t)d * 64 + e] = acc[v];
  }
}

// Reduce 8 kv partials ([d][e]) -> bf16 kvT[e][d]
__global__ __launch_bounds__(256) void kvred_kernel(
    const float* __restrict__ kvPart, bf16* __restrict__ kvT) {
  const int bh = blockIdx.x;
#pragma unroll
  for (int j = 0; j < 16; ++j) {
    const int idx = threadIdx.x + 256 * j;   // 0..4095  (= d*64+e)
    float s = 0.0f;
#pragma unroll
    for (int c = 0; c < 8; ++c) s += kvPart[((size_t)bh * 8 + c) * 4096 + idx];
    const int d = idx >> 6, e = idx & 63;
    kvT[(size_t)bh * 4096 + (size_t)e * 64 + d] = (bf16)s;
  }
}

// ---------------------------------------------------------------------------
// k_sum partials: (bh, chunk of 256 rows) -> kpart[bh][c][64]
// ---------------------------------------------------------------------------
__global__ __launch_bounds__(256) void ksum_kernel(const bf16* __restrict__ Kf,
                                                   float* __restrict__ kpart) {
  const int bh = blockIdx.x, c = blockIdx.y;
  const int b = bh >> 4, h = bh & 15;
  const int d = threadIdx.x & 63, rg = threadIdx.x >> 6;  // rg 0..3
  const bf16* p =
      Kf + ((size_t)b * 4096 + c * 256 + rg) * 1024 + (size_t)h * 64 + d;
  float s = 0.0f;
  for (int i = 0; i < 64; ++i) s += (float)p[(size_t)i * 4096];
  __shared__ float red[4][64];
  red[rg][d] = s;
  __syncthreads();
  if (threadIdx.x < 64) {
    kpart[((size_t)bh * 16 + c) * 64 + threadIdx.x] =
        red[0][threadIdx.x] + red[1][threadIdx.x] +
        red[2][threadIdx.x] + red[3][threadIdx.x];
  }
}

// ---------------------------------------------------------------------------
// denom[bh][n] = Q[n][:] . ksum[bh][:]   (ksum reduced from 16 partials)
// ---------------------------------------------------------------------------
__global__ __launch_bounds__(256) void denom_kernel(
    const bf16* __restrict__ Qf, const float* __restrict__ kpart,
    float* __restrict__ denom) {
  const int bh = blockIdx.x >> 4;
  const int chunk = blockIdx.x & 15;
  const int b = bh >> 4, h = bh & 15;
  __shared__ float ks[64];
  if (threadIdx.x < 64) {
    float s = 0.0f;
#pragma unroll
    for (int c = 0; c < 16; ++c)
      s += kpart[((size_t)bh * 16 + c) * 64 + threadIdx.x];
    ks[threadIdx.x] = s;
  }
  __syncthreads();
  const int n = chunk * 256 + threadIdx.x;
  const bf16* q = Qf + ((size_t)b * 4096 + n) * 1024 + (size_t)h * 64;
  float s = 0.0f;
#pragma unroll
  for (int d = 0; d < 64; ++d) s += (float)q[d] * ks[d];
  denom[(size_t)bh * 4096 + n] = s;
}

// ---------------------------------------------------------------------------
// num = Q * kv ; out = num / (denom + 1e-6) -> attn (bf16, [B,S,E] layout)
// grid: (S/32, B*H); 256 threads = 8 waves, tile 32(rows) x 64(cols=e).
// Entire K=64 staged once: 1 barrier, 2 WMMAs per wave.
// ---------------------------------------------------------------------------
__global__ __launch_bounds__(256) void num_kernel(
    const bf16* __restrict__ Qf, const bf16* __restrict__ kvT,
    const float* __restrict__ denom, bf16* __restrict__ attn) {
  __shared__ __align__(16) bf16 Atile[32][64];  // [m][d]  4KB
  __shared__ __align__(16) bf16 Btile[64][64];  // [e][d]  8KB
  const int bh = blockIdx.y;
  const int b = bh >> 4, h = bh & 15;
  const int rowBase = blockIdx.x * 32;
  const int tid = threadIdx.x, lane = tid & 31, wave = tid >> 5;
  const int wr = (wave >> 2) * 16;  // 0 or 16
  const int wc = (wave & 3) * 16;   // 0,16,32,48

  const int ar = tid >> 3;        // 0..31
  const int ac = (tid & 7) * 8;   // 0..56
  const int br = tid >> 2;        // 0..63
  const int bc = (tid & 3) * 16;  // 0,16,32,48

  *(uint4*)&Atile[ar][ac] = *(const uint4*)(
      Qf + ((size_t)b * 4096 + rowBase + ar) * 1024 + (size_t)h * 64 + ac);
  {
    const uint4* pb = (const uint4*)(kvT + (size_t)bh * 4096 + (size_t)br * 64 + bc);
    uint4* db = (uint4*)&Btile[br][bc];
    db[0] = pb[0];
    db[1] = pb[1];
  }
  __syncthreads();

  v8f acc = {};
#pragma unroll
  for (int kk = 0; kk < 64; kk += 32) {
    v16bf af = make_a_frag(&Atile[wr + (lane & 15)][kk], lane);
    v16bf bf = make_b_frag(&Btile[wc + (lane & 15)][kk], lane);
    acc = WMMA_BF16(af, bf, acc);
  }

  const int e = wc + (lane & 15);
#pragma unroll
  for (int v = 0; v < 8; ++v) {
    int m = wr + v + 8 * (lane >> 4);
    int n = rowBase + m;
    float den = denom[(size_t)bh * 4096 + n] + 1e-6f;
    float val = acc[v] / den;
    attn[((size_t)b * 4096 + n) * 1024 + (size_t)h * 64 + e] = (bf16)val;
  }
}

// ---------------------------------------------------------------------------
// Host launcher
// ---------------------------------------------------------------------------
extern "C" void kernel_launch(void* const* d_in, const int* in_sizes, int n_in,
                              void* d_out, int out_size, void* d_ws,
                              size_t ws_size, hipStream_t stream) {
  (void)in_sizes; (void)n_in; (void)out_size; (void)ws_size;

  const float* query    = (const float*)d_in[0];
  const float* key      = (const float*)d_in[1];
  const float* value    = (const float*)d_in[2];
  const float* q_down_w = (const float*)d_in[3];
  const float* q_up_w   = (const float*)d_in[4];
  const float* q_up_b   = (const float*)d_in[5];
  const float* k_down_w = (const float*)d_in[6];
  const float* k_up_w   = (const float*)d_in[7];
  const float* k_up_b   = (const float*)d_in[8];
  const float* v_down_w = (const float*)d_in[9];
  const float* v_up_w   = (const float*)d_in[10];
  const float* v_up_b   = (const float*)d_in[11];
  const float* out_w    = (const float*)d_in[12];
  const float* out_b    = (const float*)d_in[13];
  float* out = (float*)d_out;

  constexpr int MROWS = 4 * 4096;              // 16384
  constexpr size_t DOWN_ELEMS = (size_t)MROWS * 512;
  constexpr size_t FULL_ELEMS = (size_t)MROWS * 1024;

  char* ws = (char*)d_ws;
  size_t off = 0;
  bf16* dq = (bf16*)(ws + off); off += DOWN_ELEMS * 2;
  bf16* dk = (bf16*)(ws + off); off += DOWN_ELEMS * 2;
  bf16* dv = (bf16*)(ws + off); off += DOWN_ELEMS * 2;
  bf16* Qf = (bf16*)(ws + off); off += FULL_ELEMS * 2;
  bf16* Kf = (bf16*)(ws + off); off += FULL_ELEMS * 2;
  bf16* Vf = (bf16*)(ws + off); off += FULL_ELEMS * 2;
  bf16*  kvT    = (bf16*)(ws + off);  off += (size_t)64 * 4096 * 2;
  float* kvPart = (float*)(ws + off); off += (size_t)64 * 8 * 4096 * 4;
  float* kpart  = (float*)(ws + off); off += (size_t)64 * 16 * 64 * 4;
  float* denom  = (float*)(ws + off); off += (size_t)64 * 4096 * 4;
  // attn aliases the (by then dead) down-proj scratch: 32MB <= 48MB
  bf16* attn = dq;

  const dim3 blk(256);

  // ---- down projections: [16384,1024] x [512,1024]^T -> bf16 [16384,512] ----
  {
    dim3 grid(MROWS / 128, 512 / 64);
    gemm_wmma_kernel<true, false, 0, true><<<grid, blk, 0, stream>>>(
        query, q_down_w, nullptr, dq, MROWS, 512, 1024);
    gemm_wmma_kernel<true, false, 0, true><<<grid, blk, 0, stream>>>(
        key, k_down_w, nullptr, dk, MROWS, 512, 1024);
    gemm_wmma_kernel<true, false, 0, true><<<grid, blk, 0, stream>>>(
        value, v_down_w, nullptr, dv, MROWS, 512, 1024);
  }
  // ---- up projections (+bias, elu+1 on q/k): -> bf16 [16384,1024] ----
  {
    dim3 grid(MROWS / 128, 1024 / 64);
    gemm_wmma_kernel<false, true, 1, true><<<grid, blk, 0, stream>>>(
        dq, q_up_w, q_up_b, Qf, MROWS, 1024, 512);
    gemm_wmma_kernel<false, true, 1, true><<<grid, blk, 0, stream>>>(
        dk, k_up_w, k_up_b, Kf, MROWS, 1024, 512);
    gemm_wmma_kernel<false, true, 0, true><<<grid, blk, 0, stream>>>(
        dv, v_up_w, v_up_b, Vf, MROWS, 1024, 512);
  }
  // ---- linear-attention state ----
  ksum_kernel<<<dim3(64, 16), blk, 0, stream>>>(Kf, kpart);
  kv_kernel<<<dim3(64, 8), 512, 0, stream>>>(Kf, Vf, kvPart);
  kvred_kernel<<<64, blk, 0, stream>>>(kvPart, kvT);
  denom_kernel<<<64 * 16, blk, 0, stream>>>(Qf, kpart, denom);
  num_kernel<<<dim3(4096 / 32, 64), blk, 0, stream>>>(Qf, kvT, denom, attn);
  // ---- output projection: [16384,1024] x [1024,1024]^T + b -> f32 out ----
  {
    dim3 grid(MROWS / 128, 1024 / 64);
    gemm_wmma_kernel<false, true, 0, false><<<grid, blk, 0, stream>>>(
        attn, out_w, out_b, out, MROWS, 1024, 1024);
  }
}
